// DCNSepPreMultiOffsetFlowSimilarity_35167192220081
// MI455X (gfx1250) — matrix operand order repaired and checked by reading, not compiled, not run on hardware
//
#include <hip/hip_runtime.h>
#include <math.h>

// ---------------------------------------------------------------------------
// DCNv2 (modulated deformable conv) fused for MI455X / gfx1250.
//
// Roofline: ~7 GFLOP vs ~55 MB compulsory traffic (~2.4us @ 23.3 TB/s).
//   * never materialize sampled[B,C,K,H,W] (150 MB) -> fuse sampling into WMMA
//   * offset conv (2 GFLOP) on v_wmma (M=27 pad 32), not the fp32 VALU
//   * NCHW f32 gather -> NHWC f16 (one 32B v16h per neighbor per chunk)
//     via a streaming transpose pre-pass; bilinear blend in packed f16
//   * branch-free tap masking (clamped address + 0/1 f16 splat multiply)
// ---------------------------------------------------------------------------

typedef __attribute__((ext_vector_type(16))) _Float16 v16h;
typedef __attribute__((ext_vector_type(8)))  float    v8f;

#define Bn  4
#define Cn  64
#define On  64
#define Hn  128
#define Wn  128
#define Kp  9
#define HWn (Hn * Wn)          // 16384
#define MRM 10.0f

__device__ __forceinline__ int iclamp(int v, int lo, int hi) {
    return v < lo ? lo : (v > hi ? hi : v);
}

// ---------------------------------------------------------------------------
// K0: NCHW f32 -> NHWC f16 transpose. grid 512 blocks: first 256 = x_main,
// next 256 = x_extra. One (b,pixel) per thread: 64 coalesced b32 reads,
// 4x 32B vector stores.
// ---------------------------------------------------------------------------
__global__ __launch_bounds__(256) void nchw_to_nhwc_f16(
    const float* __restrict__ x_main, const float* __restrict__ x_extra,
    _Float16* __restrict__ xm_h, _Float16* __restrict__ xe_h)
{
    int blk = blockIdx.x;
    const float* src = (blk < 256) ? x_main : x_extra;
    _Float16*    dst = (blk < 256) ? xm_h   : xe_h;
    int p = (blk & 255) * 256 + threadIdx.x;       // b*HW + pixel, 0..65535
    int b = p >> 14, pix = p & 16383;

    const float* s = src + (size_t)b * Cn * HWn + pix;
    _Float16*    d = dst + (size_t)p * Cn;
    #pragma unroll
    for (int c0 = 0; c0 < Cn; c0 += 16) {
        v16h hv;
        #pragma unroll
        for (int j = 0; j < 16; ++j)
            hv[j] = (_Float16)s[(size_t)(c0 + j) * HWn];
        *(v16h*)(d + c0) = hv;
    }
}

// ---------------------------------------------------------------------------
// K_A: offset/mask conv as WMMA GEMM (M=27 padded to 32, K=576, N=65536),
// reading x_extra in NHWC f16 (fixed 3x3 taps, zero padding folded into a
// 0/1 f16 mask on a clamped address -> branch-free), then per-pixel
// tanh/sigmoid epilogue via a per-wave LDS D-tile transpose.
// Writes py/px/mask fields [B,K,H,W] f32.
// ---------------------------------------------------------------------------
__global__ __launch_bounds__(256) void dcn_offset_wmma_kernel(
    const _Float16* __restrict__ xe_h,      // [B,H,W,64] f16
    const float* __restrict__ pre_offset,   // [B,K,H,W,2] (x,y)
    const float* __restrict__ pre_sim,      // [B,K,H,W]
    const float* __restrict__ w_om,         // [27,64,3,3]
    const float* __restrict__ b_om,         // [27]
    float* __restrict__ ws_py,
    float* __restrict__ ws_px,
    float* __restrict__ ws_m)
{
    extern __shared__ char smA[];
    v16h*  fragW = (v16h*)smA;                    // 18 chunks * 2 mtiles * 32
    float* eps   = (float*)(smA + 18 * 2 * 32 * sizeof(v16h)); // 8 waves * 512

    const int tid  = threadIdx.x;
    const int lane = tid & 31;
    const int wave = tid >> 5;

    // stage w_om -> f16 A-fragments (rows 27..31 zero-padded)
    for (int e = tid; e < 18 * 2 * 32; e += 256) {
        int l     = e & 31;
        int f     = e >> 5;
        int mt    = f & 1;
        int chunk = f >> 1;
        int j     = mt * 16 + (l & 15);
        int khalf = (l & 16) ? 8 : 0;
        v16h hv;
        #pragma unroll
        for (int i = 0; i < 16; ++i) {
            int kloc = 2 * khalf + i;
            int ck   = chunk * 32 + kloc;   // ck = ktap*64 + c
            int kf   = ck >> 6;
            int c    = ck & 63;
            hv[i] = (j < 27) ? (_Float16)w_om[((size_t)j * Cn + c) * 9 + kf]
                             : (_Float16)0.0f;
        }
        fragW[e] = hv;
    }
    __syncthreads();

    const int tile  = blockIdx.x * 8 + wave;
    const int b     = tile >> 10;
    const int pix0  = (tile & 1023) * 16;
    const int pixel = pix0 + (lane & 15);
    const int h     = pixel >> 7;
    const int w     = pixel & 127;
    const int chalf = (lane & 16) ? 16 : 0;
    const _Float16* img = xe_h + (size_t)b * HWn * Cn;

    v8f acc[2] = {};

    for (int k = 0; k < Kp; ++k) {
        int sy = h + k / 3 - 1;
        int sx = w + k % 3 - 1;
        _Float16 vmask = (sy >= 0 && sy < Hn && sx >= 0 && sx < Wn)
                         ? (_Float16)1.0f : (_Float16)0.0f;
        int syc = iclamp(sy, 0, Hn - 1);
        int sxc = iclamp(sx, 0, Wn - 1);
        const _Float16* tap = img + (size_t)(syc * Wn + sxc) * Cn;
        #pragma unroll
        for (int ch2 = 0; ch2 < 2; ++ch2) {
            int cstart = ch2 * 32 + chalf;
            v16h bfrag = (*(const v16h*)(tap + cstart)) * vmask;
            int chunk = k * 2 + ch2;
            #pragma unroll
            for (int mt = 0; mt < 2; ++mt) {
                v16h a = fragW[(chunk * 2 + mt) * 32 + lane];
                acc[mt] = __builtin_amdgcn_wmma_f32_16x16x32_f16(
                    false, a, false, bfrag, (short)0, acc[mt], false, false);
            }
        }
    }

    // D tile -> per-wave LDS so each lane can see all 27 channels of a pixel
    const int mrow = (lane & 16) ? 8 : 0;
    float* ep = eps + wave * 512;
    #pragma unroll
    for (int mt = 0; mt < 2; ++mt)
        #pragma unroll
        for (int r = 0; r < 8; ++r)
            ep[(mt * 16 + mrow + r) * 16 + (lane & 15)] = acc[mt][r];
    __syncthreads();

    if (lane < 16) {
        const int pix = lane;
        #pragma unroll
        for (int k = 0; k < Kp; ++k) {
            float cy = ep[(2 * k) * 16 + pix]     + b_om[2 * k];
            float cx = ep[(2 * k + 1) * 16 + pix] + b_om[2 * k + 1];
            float cm = ep[(18 + k) * 16 + pix]    + b_om[18 + k];
            size_t idx = (size_t)(b * Kp + k) * HWn + pixel;
            float offy = MRM * tanhf(cy);
            float offx = MRM * tanhf(cx);
            float m    = 1.0f / (1.0f + expf(-cm * pre_sim[idx]));
            float prex = pre_offset[idx * 2 + 0];
            float prey = pre_offset[idx * 2 + 1];
            ws_py[idx] = offy + prey + (float)(h - 1 + k / 3);
            ws_px[idx] = offx + prex + (float)(w - 1 + k % 3);
            ws_m[idx]  = m;
        }
    }
}

// ---------------------------------------------------------------------------
// K_B: fused bilinear-sample + WMMA GEMM (M=64, K=576, N=65536).
// NHWC f16 gather: one v16h (32B) load per neighbor per chunk; bilinear
// blend + mask folded into packed-f16 weights (v_pk_fma_f16).
// ---------------------------------------------------------------------------
__global__ __launch_bounds__(256) void dcn_wmma_kernel(
    const _Float16* __restrict__ xm_h,      // [B,H,W,64] f16
    const float* __restrict__ weight,       // [64,64,3,3]
    const float* __restrict__ bias,         // [64]
    const float* __restrict__ ws_py,
    const float* __restrict__ ws_px,
    const float* __restrict__ ws_m,
    float* __restrict__ out)                // [B,64,H,W]
{
    extern __shared__ v16h fragA[];         // 18 chunks * 4 mtiles * 32 lanes

    const int tid  = threadIdx.x;
    const int lane = tid & 31;
    const int wave = tid >> 5;

    for (int e = tid; e < 18 * 4 * 32; e += 256) {
        int l     = e & 31;
        int f     = e >> 5;
        int mt    = f & 3;
        int chunk = f >> 2;
        int o     = mt * 16 + (l & 15);
        int khalf = (l & 16) ? 8 : 0;
        v16h hv;
        #pragma unroll
        for (int i = 0; i < 16; ++i) {
            int kloc = 2 * khalf + i;
            int ck   = chunk * 32 + kloc;   // ck = ktap*64 + c
            int kf   = ck >> 6;
            int c    = ck & 63;
            hv[i] = (_Float16)weight[((size_t)o * Cn + c) * 9 + kf];
        }
        fragA[e] = hv;
    }
    __syncthreads();

    const int tile  = blockIdx.x * 8 + wave;
    const int b     = tile >> 10;
    const int pix0  = (tile & 1023) * 16;
    const int pixel = pix0 + (lane & 15);
    const int chalf = (lane & 16) ? 16 : 0;
    const _Float16* img = xm_h + (size_t)b * HWn * Cn;

    v8f acc[4] = {};

    for (int k = 0; k < Kp; ++k) {
        size_t widx = (size_t)(b * Kp + k) * HWn + pixel;
        float py = ws_py[widx];
        float px = ws_px[widx];
        float m  = ws_m[widx];

        float fy = floorf(py), fx = floorf(px);
        float dy = py - fy,    dx = px - fx;
        int y0 = (int)fy, x0 = (int)fx;
        int y1 = y0 + 1,  x1 = x0 + 1;

        float vy0 = (y0 >= 0 && y0 < Hn) ? 1.0f : 0.0f;
        float vy1 = (y1 >= 0 && y1 < Hn) ? 1.0f : 0.0f;
        float vx0 = (x0 >= 0 && x0 < Wn) ? 1.0f : 0.0f;
        float vx1 = (x1 >= 0 && x1 < Wn) ? 1.0f : 0.0f;
        _Float16 h00 = (_Float16)((1.0f - dy) * (1.0f - dx) * m * vy0 * vx0);
        _Float16 h01 = (_Float16)((1.0f - dy) * dx          * m * vy0 * vx1);
        _Float16 h10 = (_Float16)(dy * (1.0f - dx)          * m * vy1 * vx0);
        _Float16 h11 = (_Float16)(dy * dx                   * m * vy1 * vx1);

        int y0c = iclamp(y0, 0, Hn - 1), y1c = iclamp(y1, 0, Hn - 1);
        int x0c = iclamp(x0, 0, Wn - 1), x1c = iclamp(x1, 0, Wn - 1);
        const _Float16* p00 = img + (size_t)(y0c * Wn + x0c) * Cn;
        const _Float16* p01 = img + (size_t)(y0c * Wn + x1c) * Cn;
        const _Float16* p10 = img + (size_t)(y1c * Wn + x0c) * Cn;
        const _Float16* p11 = img + (size_t)(y1c * Wn + x1c) * Cn;

        #pragma unroll
        for (int ch2 = 0; ch2 < 2; ++ch2) {
            int cstart = ch2 * 32 + chalf;
            v16h n00 = *(const v16h*)(p00 + cstart);
            v16h n01 = *(const v16h*)(p01 + cstart);
            v16h n10 = *(const v16h*)(p10 + cstart);
            v16h n11 = *(const v16h*)(p11 + cstart);
            v16h bfrag = n00 * h00;
            bfrag += n01 * h01;
            bfrag += n10 * h10;
            bfrag += n11 * h11;

            int chunk = k * 2 + ch2;
            #pragma unroll
            for (int mt = 0; mt < 4; ++mt) {
                v16h a = fragA[(chunk * 4 + mt) * 32 + lane];
                acc[mt] = __builtin_amdgcn_wmma_f32_16x16x32_f16(
                    false, a, false, bfrag, (short)0, acc[mt], false, false);
            }
        }
    }

    const int mrow = (lane & 16) ? 8 : 0;
    #pragma unroll
    for (int mt = 0; mt < 4; ++mt) {
        #pragma unroll
        for (int r = 0; r < 8; ++r) {
            int o = mt * 16 + mrow + r;
            out[(size_t)(b * On + o) * HWn + pixel] = acc[mt][r] + bias[o];
        }
    }
}

// ---------------------------------------------------------------------------
// Launch. Workspace layout (23.9 MB total):
//   [0)            xm_h : B*HW*64 f16  = 8,388,608 B
//   [8,388,608)    xe_h : B*HW*64 f16  = 8,388,608 B
//   [16,777,216)   py/px/m fields: 3 * B*K*HW f32 = 7,077,888 B
// Input order: x_main, x_extra, pre_offset, pre_sim, weight, bias, w_om, b_om.
// ---------------------------------------------------------------------------
extern "C" void kernel_launch(void* const* d_in, const int* in_sizes, int n_in,
                              void* d_out, int out_size, void* d_ws, size_t ws_size,
                              hipStream_t stream) {
    (void)in_sizes; (void)n_in; (void)out_size; (void)ws_size;

    const float* x_main     = (const float*)d_in[0];
    const float* x_extra    = (const float*)d_in[1];
    const float* pre_offset = (const float*)d_in[2];
    const float* pre_sim    = (const float*)d_in[3];
    const float* weight     = (const float*)d_in[4];
    const float* bias       = (const float*)d_in[5];
    const float* w_om       = (const float*)d_in[6];
    const float* b_om       = (const float*)d_in[7];
    float* out = (float*)d_out;

    const size_t NC   = (size_t)Bn * HWn * Cn;      // 4,194,304 elements
    const size_t BKHW = (size_t)Bn * Kp * HWn;      // 589,824 elements
    _Float16* xm_h = (_Float16*)d_ws;
    _Float16* xe_h = xm_h + NC;
    float* ws_py = (float*)(xe_h + NC);
    float* ws_px = ws_py + BKHW;
    float* ws_m  = ws_px + BKHW;

    // K0: NCHW f32 -> NHWC f16 for both images (512 blocks)
    nchw_to_nhwc_f16<<<512, 256, 0, stream>>>(x_main, x_extra, xm_h, xe_h);

    // K_A: offset conv on WMMA; dyn LDS = 36,864 (fragments) + 16,384 (D tiles)
    dcn_offset_wmma_kernel<<<512, 256, 18 * 2 * 32 * (int)sizeof(v16h) + 8 * 512 * (int)sizeof(float), stream>>>(
        xe_h, pre_offset, pre_sim, w_om, b_om, ws_py, ws_px, ws_m);

    // K_B: fused sample+GEMM; dyn LDS = 73,728 B A-fragments
    dcn_wmma_kernel<<<512, 256, 18 * 4 * 32 * (int)sizeof(v16h), stream>>>(
        xm_h, weight, bias, ws_py, ws_px, ws_m, out);
}
